// LiteTransformer_25915832664792
// MI455X (gfx1250) — matrix operand order repaired and checked by reading, not compiled, not run on hardware
//
#include <hip/hip_runtime.h>
#include <math.h>

// ---------------------------------------------------------------------------
// LiteTransformer for MI455X (gfx1250): bf16 WMMA GEMMs, double-buffered async
// global->LDS staging of B tiles, VALU pos-attention with LDS radix-select.
// B=4, RES=64, N=4096, M=1024, H=8, D=256, K=32, NB=4
// ---------------------------------------------------------------------------

#define BB 4
#define NN 4096
#define MM 1024
#define DD 256
#define KH 32
#define HH 8
#define NB 4

typedef __attribute__((ext_vector_type(16))) __bf16 v16bf;
typedef __attribute__((ext_vector_type(16))) short  v16s;
typedef __attribute__((ext_vector_type(8)))  float  v8f;

__device__ __forceinline__ unsigned short f2bf(float f) {
  unsigned u = __float_as_uint(f);
  unsigned r = u + 0x7FFFu + ((u >> 16) & 1u);   // round-to-nearest-even
  return (unsigned short)(r >> 16);
}

__device__ __forceinline__ float gelu_f(float x) {
  return 0.5f * x * (1.0f + erff(x * 0.70710678118654752f));
}

__device__ __forceinline__ unsigned f2ord(float f) {
  unsigned u = __float_as_uint(f);
  return (u & 0x80000000u) ? ~u : (u | 0x80000000u);
}
__device__ __forceinline__ float ord2f(unsigned u) {
  unsigned b = (u & 0x80000000u) ? (u ^ 0x80000000u) : ~u;
  return __uint_as_float(b);
}

// ---------------------------------------------------------------------------
// Batched GEMM: C = act(scale * A(bf16) @ Bt^T(bf16) + bias + resid)
//  - A  : (rows x Kd) row-major bf16, lda row stride
//  - Bt : (Ncols x Kd) row-major bf16 (B transposed, k contiguous)
//  - 256 threads = 8 waves; workgroup = 128 rows x (NSUB*16) cols of C.
//  - Double-buffered LDS staging: per 32-k step one async b128 copy per
//    thread, s_wait_asynccnt + barrier, next tile's copy issued into the
//    other buffer before computing (copy overlaps WMMA of previous step).
//  - NSUB is compile-time so the WMMA loop has no guards / no acc shuffles.
// Fragment layouts (A/B: this code's contiguous k order, not the ISA table order; C per ISA):
//  A 16x32: lane l (m=l&15), slot hh -> k = hh + 16*(l>>4)
//  B 32x16: lane l (n=l&15), slot hh -> k = hh + 16*(l>>4)
//  C 16x16: lane l (n=l&15), vgpr r  -> m = r + 8*(l>>4)
// ---------------------------------------------------------------------------
template <int NSUB>
__global__ __launch_bounds__(256) void gemm_wmma_bf16(
    const unsigned short* __restrict__ A,  long lda, long sAz,
    const unsigned short* __restrict__ Bt, long ldb, long sBz,
    float* __restrict__ Cf, unsigned short* __restrict__ Cb,
    long ldc, long sCz, int cTrans,
    const float* __restrict__ bias,
    const float* __restrict__ resid, long ldr,
    int Kd, float scale, int act)
{
  // single shared object -> LDS offset 0; two [n][k] tiles of 64 rows x 32 bf16
  __shared__ __align__(32) unsigned short sB[2][64 * 32];

  const unsigned short* Ab  = A  + (long)blockIdx.z * sAz;
  const unsigned short* Btb = Bt + (long)blockIdx.z * sBz;
  const long cofs = (long)blockIdx.z * sCz;

  const int tid    = threadIdx.x;
  const int lane   = tid & 31;
  const int wave   = tid >> 5;
  const int lhalf  = lane >> 4;
  const int lmod   = lane & 15;
  const int mrow   = blockIdx.y * 128 + wave * 16 + lmod;
  const int nbase0 = blockIdx.x * (NSUB * 16);

  // staging assignment: thread copies 16 bytes: row sn, 8-element chunk sc
  const int  sn     = tid >> 2;
  const int  sc     = tid & 3;
  const bool doCopy = (NSUB == 4) || (sn < NSUB * 16);
  const int  snc    = doCopy ? sn : 0;
  const unsigned short* gsrcBase = Btb + (long)(nbase0 + snc) * ldb + sc * 8;
  const unsigned ldsRow = (unsigned)(snc * 64 + sc * 16);   // bytes within a tile

  // prologue: stage k0=0 tile into buffer 0
  if (doCopy) {
    const unsigned long long g = (unsigned long long)(size_t)gsrcBase;
    asm volatile("global_load_async_to_lds_b128 %0, %1, off"
                 :: "v"(ldsRow), "v"(g) : "memory");
  }

  v8f acc[NSUB];
#pragma unroll
  for (int j = 0; j < NSUB; ++j)
#pragma unroll
    for (int r = 0; r < 8; ++r) acc[j][r] = 0.0f;

  const unsigned short* Arow = Ab + (long)mrow * lda;

  int buf = 0;
  for (int k0 = 0; k0 < Kd; k0 += 32) {
    asm volatile("s_wait_asynccnt 0x0" ::: "memory");
    __syncthreads();   // current tile visible; previous tile's readers done

    if ((k0 + 32 < Kd) && doCopy) {   // refill the other buffer (overlapped)
      const unsigned ldsOff = (unsigned)((buf ^ 1) * 4096) + ldsRow;
      const unsigned long long g = (unsigned long long)(size_t)(gsrcBase + k0 + 32);
      asm volatile("global_load_async_to_lds_b128 %0, %1, off"
                   :: "v"(ldsOff), "v"(g) : "memory");
    }

    // A fragment straight from global (vectorizes to 2x b128)
    v16s av;
#pragma unroll
    for (int hh = 0; hh < 16; ++hh) {
      const int kk = k0 + hh + 16 * lhalf;
      av[hh] = (short)Arow[kk];
    }
    const v16bf aV = __builtin_bit_cast(v16bf, av);

    const unsigned short* sbuf = &sB[buf][0];
#pragma unroll
    for (int j = 0; j < NSUB; ++j) {
      const v16s bv = *(const v16s*)&sbuf[(j * 16 + lmod) * 32 + lhalf * 16];
      acc[j] = __builtin_amdgcn_wmma_f32_16x16x32_bf16(
          false, aV, false, __builtin_bit_cast(v16bf, bv),
          (short)0, acc[j], false, false);
    }
    buf ^= 1;
  }

  // ---- epilogue (tiles are exact; no guards) ----
#pragma unroll
  for (int j = 0; j < NSUB; ++j) {
    const int   n  = nbase0 + j * 16 + lmod;
    const float bv = bias ? bias[n] : 0.0f;
#pragma unroll
    for (int r = 0; r < 8; ++r) {
      const int m = mrow - lmod + r + 8 * lhalf;
      float v = acc[j][r] * scale + bv;
      if (resid) v += resid[(long)m * ldr + n];
      if (act)   v  = gelu_f(v);
      const long ci = cofs + (cTrans ? ((long)n * ldc + m) : ((long)m * ldc + n));
      if (Cf) Cf[ci] = v;
      if (Cb) Cb[ci] = f2bf(v);
    }
  }
}

// ---------------------------------------------------------------------------
// Encoder: en[bn,d] = gelu(gx*Wen[0,d] + gy*Wen[1,d] + inp[bn]*Wen[2,d] + b[d])
// ---------------------------------------------------------------------------
__global__ void encode_k(const float* __restrict__ inp, const float* __restrict__ Wen,
                         const float* __restrict__ ben, unsigned short* __restrict__ en_bf)
{
  const int idx = blockIdx.x * 256 + threadIdx.x;      // B*N*D threads
  const int d  = idx & 255;
  const int bn = idx >> 8;
  const int n  = bn & (NN - 1);
  const int ix = n >> 6, iy = n & 63;
  const float fx = (float)ix * (1.0f / 64.0f);
  const float fy = (float)iy * (1.0f / 64.0f);
  const float v  = inp[bn];
  const float e  = fx * Wen[d] + fy * Wen[256 + d] + v * Wen[512 + d] + ben[d];
  en_bf[idx] = f2bf(gelu_f(e));
}

// per-head positional scale: tan(pi/4 * (1 + sin(r)))
__global__ void head_scales_k(const float* __restrict__ rdn, const float* __restrict__ rup,
                              float* __restrict__ hs)
{
  const int t = threadIdx.x;
  if (t < 8)       hs[t] = tanf(0.78539816339744831f * (1.0f + sinf(rdn[t])));
  else if (t < 16) hs[t] = tanf(0.78539816339744831f * (1.0f + sinf(rup[t - 8])));
}

__global__ void f32_to_bf16_k(const float* __restrict__ in, unsigned short* __restrict__ out, int n)
{
  const int i = blockIdx.x * 256 + threadIdx.x;
  if (i < n) out[i] = f2bf(in[i]);
}

// (H,D,K) f32 -> bf16 transposed weight Wt[(h*32+k)*256 + d]
__global__ void conv_hdk_k(const float* __restrict__ w, unsigned short* __restrict__ out)
{
  const int i = blockIdx.x * 256 + threadIdx.x;        // H*D*K = 65536
  const int h = i >> 13;
  const int d = (i >> 5) & 255;
  const int k = i & 31;
  out[(h * 32 + k) * 256 + d] = f2bf(w[i]);
}

// slabs of 256x256 f32 (in,out) -> bf16 transposed (out,in) per slab
__global__ void conv_sqT_k(const float* __restrict__ w, unsigned short* __restrict__ out, int total)
{
  const int i = blockIdx.x * 256 + threadIdx.x;
  if (i < total) {
    const int slab = i >> 16;
    const int r = i & 65535;
    const int d = r >> 8;
    const int n = r & 255;
    out[slab * 65536 + n * 256 + d] = f2bf(w[i]);
  }
}

// ---------------------------------------------------------------------------
// Row softmax: S (rows x ncols, f32) -> P (bf16)
// ---------------------------------------------------------------------------
__global__ void softmax_rows_k(const float* __restrict__ S, unsigned short* __restrict__ P, int ncols)
{
  __shared__ float sred[256];
  const long row = blockIdx.x;
  const int  tid = threadIdx.x;
  const float* sp = S + row * ncols;

  float lmax = -3.4e38f;
  for (int i = tid; i < ncols; i += 256) lmax = fmaxf(lmax, sp[i]);
  sred[tid] = lmax; __syncthreads();
  for (int off = 128; off > 0; off >>= 1) {
    if (tid < off) sred[tid] = fmaxf(sred[tid], sred[tid + off]);
    __syncthreads();
  }
  const float mx = sred[0]; __syncthreads();

  float lsum = 0.0f;
  for (int i = tid; i < ncols; i += 256) lsum += expf(sp[i] - mx);
  sred[tid] = lsum; __syncthreads();
  for (int off = 128; off > 0; off >>= 1) {
    if (tid < off) sred[tid] += sred[tid + off];
    __syncthreads();
  }
  const float inv = 1.0f / sred[0]; __syncthreads();

  for (int i = tid; i < ncols; i += 256)
    P[row * ncols + i] = f2bf(expf(sp[i] - mx) * inv);
}

// ---------------------------------------------------------------------------
// LDS radix select: k-th smallest (0-indexed) of sdat[0..n)
// ---------------------------------------------------------------------------
__device__ float radix_select(const float* sdat, int n, int k,
                              unsigned* hist, unsigned* s_prefix, int* s_k)
{
  const int tid = threadIdx.x;
  if (tid == 0) { *s_prefix = 0u; *s_k = k; }
  __syncthreads();
  for (int pass = 0; pass < 4; ++pass) {
    const int shift = 24 - 8 * pass;
    const unsigned hmask = (pass == 0) ? 0u : (0xFFFFFFFFu << (shift + 8));
    for (int i = tid; i < 256; i += 256) hist[i] = 0u;
    __syncthreads();
    const unsigned pref = *s_prefix;
    for (int i = tid; i < n; i += 256) {
      const unsigned u = f2ord(sdat[i]);
      if ((u & hmask) == pref) atomicAdd(&hist[(u >> shift) & 0xFFu], 1u);
    }
    __syncthreads();
    if (tid == 0) {
      int kk = *s_k; unsigned cum = 0; int bin = 0;
      for (bin = 0; bin < 256; ++bin) {
        const unsigned c = hist[bin];
        if (cum + c > (unsigned)kk) break;
        cum += c;
      }
      *s_prefix = pref | ((unsigned)bin << shift);
      *s_k = kk - (int)cum;
    }
    __syncthreads();
  }
  const unsigned res = *s_prefix;
  __syncthreads();
  return ord2f(res);
}

// ---------------------------------------------------------------------------
// Positional attention with per-row linear-quantile threshold.
// grid = (rows_out, H), 256 threads. dist(row, i) = mcross[row*rowMul + i*nMul]
// out[b, row, h*32+k] = gelu( sum_i att[i] * val[b, i, h*32+k] )
// ---------------------------------------------------------------------------
__global__ void pos_att_k(const float* __restrict__ mcross,
                          const float* __restrict__ hscale, int hsOff,
                          const float* __restrict__ val, float* __restrict__ out,
                          int n_in, long rowMul, long nMul,
                          int rank, float frac,
                          long valBatchStride, long outBatchStride)
{
  __shared__ float    satt[4096];
  __shared__ unsigned hist[256];
  __shared__ float    sred[256];
  __shared__ unsigned s_prefix;
  __shared__ int      s_k;
  __shared__ float    s_thr, s_minv, s_sum;

  const int row = blockIdx.x;
  const int h   = blockIdx.y;
  const int tid = threadIdx.x;
  const float hs = hscale[hsOff + h];

  float lmin = 3.4e38f;
  for (int i = tid; i < n_in; i += 256) {
    const float s = mcross[(long)row * rowMul + (long)i * nMul] * hs;
    satt[i] = s;
    lmin = fminf(lmin, s);
  }
  sred[tid] = lmin; __syncthreads();
  for (int off = 128; off > 0; off >>= 1) {
    if (tid < off) sred[tid] = fminf(sred[tid], sred[tid + off]);
    __syncthreads();
  }
  if (tid == 0) s_minv = sred[0];
  __syncthreads();

  const float a  = radix_select(satt, n_in, rank,     hist, &s_prefix, &s_k);
  const float b2 = radix_select(satt, n_in, rank + 1, hist, &s_prefix, &s_k);
  if (tid == 0) s_thr = a + frac * (b2 - a);
  __syncthreads();

  const float thr = s_thr, minv = s_minv;
  float lsum = 0.0f;
  for (int i = tid; i < n_in; i += 256) {
    const float s = satt[i];
    const float w = (s <= thr) ? expf(minv - s) : 0.0f;   // softmax(-scaled), masked
    satt[i] = w;
    lsum += w;
  }
  sred[tid] = lsum; __syncthreads();
  for (int off = 128; off > 0; off >>= 1) {
    if (tid < off) sred[tid] += sred[tid + off];
    __syncthreads();
  }
  if (tid == 0) s_sum = sred[0];
  __syncthreads();
  const float inv = 1.0f / s_sum;

  // weighted sum over n_in: 128 (b,k) pairs, split the n-range in two halves
  const int pair = tid & 127;
  const int half = tid >> 7;
  const int b    = pair >> 5;
  const int kk   = pair & 31;
  const int nh   = n_in >> 1;
  const float* vp = val + (long)b * valBatchStride + (long)h * 32 + kk;
  float acc = 0.0f;
  const int n0 = half * nh;
  for (int n = n0; n < n0 + nh; ++n) acc += satt[n] * vp[(long)n * 256];
  sred[tid] = acc; __syncthreads();
  if (tid < 128) {
    const float tot = (sred[tid] + sred[tid + 128]) * inv;
    out[(long)b * outBatchStride + (long)row * 256 + h * 32 + kk] = gelu_f(tot);
  }
}

// final projection to OUT=1 plus de-normalization
__global__ void decode_final_k(const float* __restrict__ t, const float* __restrict__ wde2,
                               const float* __restrict__ bde2, const float* __restrict__ ymean,
                               const float* __restrict__ ystd, float* __restrict__ out)
{
  const int bn = blockIdx.x * 256 + threadIdx.x;       // B*N threads
  const float* tp = t + (long)bn * 256;
  float s = 0.0f;
#pragma unroll 8
  for (int d = 0; d < 256; ++d) s += tp[d] * wde2[d];
  out[bn] = (s + bde2[0]) * ystd[0] + ymean[0];
}

// ---------------------------------------------------------------------------
// Host-side orchestration
// ---------------------------------------------------------------------------
extern "C" void kernel_launch(void* const* d_in, const int* in_sizes, int n_in,
                              void* d_out, int out_size, void* d_ws, size_t ws_size,
                              hipStream_t stream)
{
  (void)in_sizes; (void)n_in; (void)out_size; (void)ws_size;

  const float* inp     = (const float*)d_in[0];
  const float* m_cross = (const float*)d_in[1];
  const float* W_en    = (const float*)d_in[2];
  const float* b_en    = (const float*)d_in[3];
  const float* r_down  = (const float*)d_in[4];
  const float* w_down  = (const float*)d_in[5];
  const float* q_pa    = (const float*)d_in[6];
  const float* k_pa    = (const float*)d_in[7];
  const float* v_pa    = (const float*)d_in[8];
  const float* W1_mlp  = (const float*)d_in[9];
  const float* b1_mlp  = (const float*)d_in[10];
  const float* W2_mlp  = (const float*)d_in[11];
  const float* b2_mlp  = (const float*)d_in[12];
  const float* W_res   = (const float*)d_in[13];
  const float* b_res   = (const float*)d_in[14];
  const float* r_up    = (const float*)d_in[15];
  const float* w_up    = (const float*)d_in[16];
  const float* W_de1   = (const float*)d_in[17];
  const float* b_de1   = (const float*)d_in[18];
  const float* W_de2   = (const float*)d_in[19];
  const float* b_de2   = (const float*)d_in[20];
  const float* y_mean  = (const float*)d_in[21];
  const float* y_std   = (const float*)d_in[22];
  float* outp = (float*)d_out;

  // bump allocator over workspace
  char* wp = (char*)d_ws;
  auto alloc = [&](size_t bytes) -> void* {
    void* p = (void*)wp;
    wp += (bytes + 255) & ~(size_t)255;
    return p;
  };

  float*          hs     = (float*)         alloc(16 * sizeof(float));
  unsigned short* en_bf  = (unsigned short*)alloc((size_t)BB * NN * DD * 2);
  float*          val_dn = (float*)         alloc((size_t)BB * NN * DD * 4);
  float*          x_f    = (float*)         alloc((size_t)BB * MM * DD * 4);
  unsigned short* x_bf   = (unsigned short*)alloc((size_t)BB * MM * DD * 2);
  unsigned short* q_bf   = (unsigned short*)alloc((size_t)BB * MM * DD * 2);
  unsigned short* kk_bf  = (unsigned short*)alloc((size_t)BB * MM * DD * 2);
  unsigned short* vt_bf  = (unsigned short*)alloc((size_t)BB * DD * MM * 2);  // V^T (B,D,M)
  float*          Smat   = (float*)         alloc((size_t)HH * MM * MM * 4);
  unsigned short* P_bf   = (unsigned short*)alloc((size_t)HH * MM * MM * 2);
  float*          pa_f   = (float*)         alloc((size_t)BB * MM * DD * 4);
  unsigned short* pa_bf  = (unsigned short*)alloc((size_t)BB * MM * DD * 2);
  float*          t1_f   = (float*)         alloc((size_t)BB * MM * DD * 4);
  unsigned short* t1_bf  = (unsigned short*)alloc((size_t)BB * MM * DD * 2);
  float*          res_f  = (float*)         alloc((size_t)BB * MM * DD * 4);
  float*          val_up = (float*)         alloc((size_t)BB * MM * DD * 4);
  float*          de_f   = (float*)         alloc((size_t)BB * NN * DD * 4);
  unsigned short* de_bf  = (unsigned short*)alloc((size_t)BB * NN * DD * 2);
  float*          tde_f  = (float*)         alloc((size_t)BB * NN * DD * 4);
  unsigned short* wdn_bf = (unsigned short*)alloc((size_t)DD * DD * 2);
  unsigned short* wup_bf = (unsigned short*)alloc((size_t)DD * DD * 2);
  unsigned short* wq_bf  = (unsigned short*)alloc((size_t)NB * DD * DD * 2);
  unsigned short* wk_bf  = (unsigned short*)alloc((size_t)NB * DD * DD * 2);
  unsigned short* wv_bf  = (unsigned short*)alloc((size_t)NB * DD * DD * 2);
  unsigned short* w1_bf  = (unsigned short*)alloc((size_t)NB * DD * DD * 2);
  unsigned short* w2_bf  = (unsigned short*)alloc((size_t)NB * DD * DD * 2);
  unsigned short* wr_bf  = (unsigned short*)alloc((size_t)NB * DD * DD * 2);
  unsigned short* wd1_bf = (unsigned short*)alloc((size_t)DD * DD * 2);

  const long DDsq = (long)DD * DD;   // 65536

  // --- setup: scales, encode, weight repacks / conversions (all transposed) ---
  head_scales_k<<<1, 32, 0, stream>>>(r_down, r_up, hs);
  encode_k<<<(BB * NN * DD) / 256, 256, 0, stream>>>(inp, W_en, b_en, en_bf);

  conv_hdk_k<<<DDsq / 256, 256, 0, stream>>>(w_down, wdn_bf);
  conv_hdk_k<<<DDsq / 256, 256, 0, stream>>>(w_up, wup_bf);
  for (int i = 0; i < NB; ++i) {
    conv_hdk_k<<<DDsq / 256, 256, 0, stream>>>(q_pa + i * DDsq, wq_bf + i * DDsq);
    conv_hdk_k<<<DDsq / 256, 256, 0, stream>>>(k_pa + i * DDsq, wk_bf + i * DDsq);
    conv_hdk_k<<<DDsq / 256, 256, 0, stream>>>(v_pa + i * DDsq, wv_bf + i * DDsq);
  }
  conv_sqT_k<<<(NB * DDsq) / 256, 256, 0, stream>>>(W1_mlp, w1_bf, NB * DDsq);
  conv_sqT_k<<<(NB * DDsq) / 256, 256, 0, stream>>>(W2_mlp, w2_bf, NB * DDsq);
  conv_sqT_k<<<(NB * DDsq) / 256, 256, 0, stream>>>(W_res,  wr_bf, NB * DDsq);
  conv_sqT_k<<<DDsq / 256, 256, 0, stream>>>(W_de1, wd1_bf, (int)DDsq);

  // --- val_down = en @ w_down : (B*N x 256) ---
  gemm_wmma_bf16<4><<<dim3(4, (BB * NN) / 128, 1), 256, 0, stream>>>(
      en_bf, DD, 0, wdn_bf, DD, 0,
      val_dn, nullptr, DD, 0, 0, nullptr, nullptr, 0, DD, 1.0f, 0);

  // --- pos-att down: N=4096 in, M=1024 out; rank 1228, frac 0.5 ---
  pos_att_k<<<dim3(MM, HH), 256, 0, stream>>>(
      m_cross, hs, 0, val_dn, x_f,
      NN, 1L, (long)MM, 1228, 0.5f,
      (long)NN * DD, (long)MM * DD);

  // --- transformer blocks ---
  const float sscale = 0.17677669529663689f;   // 1/sqrt(32)
  for (int i = 0; i < NB; ++i) {
    f32_to_bf16_k<<<(BB * MM * DD) / 256, 256, 0, stream>>>(x_f, x_bf, BB * MM * DD);

    // Q, K projections -> bf16 (token-major); V -> bf16 transposed (dim-major)
    gemm_wmma_bf16<4><<<dim3(4, (BB * MM) / 128, 1), 256, 0, stream>>>(
        x_bf, DD, 0, wq_bf + i * DDsq, DD, 0,
        nullptr, q_bf, DD, 0, 0, nullptr, nullptr, 0, DD, 1.0f, 0);
    gemm_wmma_bf16<4><<<dim3(4, (BB * MM) / 128, 1), 256, 0, stream>>>(
        x_bf, DD, 0, wk_bf + i * DDsq, DD, 0,
        nullptr, kk_bf, DD, 0, 0, nullptr, nullptr, 0, DD, 1.0f, 0);
    for (int b = 0; b < BB; ++b) {
      gemm_wmma_bf16<4><<<dim3(4, MM / 128, 1), 256, 0, stream>>>(
          x_bf + (long)b * MM * DD, DD, 0, wv_bf + i * DDsq, DD, 0,
          nullptr, vt_bf + (long)b * DD * MM, MM, 0, 1 /*cTrans*/,
          nullptr, nullptr, 0, DD, 1.0f, 0);
    }

    // attention per batch, batched over heads in grid.z
    for (int b = 0; b < BB; ++b) {
      const long boff = (long)b * MM * DD;
      // S = (Q K^T) / sqrt(32): Bt = K (token-major, k contiguous), Kd=32
      gemm_wmma_bf16<4><<<dim3(16, MM / 128, HH), 256, 0, stream>>>(
          q_bf + boff, DD, 32, kk_bf + boff, DD, 32,
          Smat, nullptr, MM, (long)MM * MM, 0,
          nullptr, nullptr, 0, KH, sscale, 0);
      softmax_rows_k<<<HH * MM, 256, 0, stream>>>(Smat, P_bf, MM);
      // O = P V: Bt = V^T slice (32 x 1024, k=token contiguous), gelu epilogue
      gemm_wmma_bf16<2><<<dim3(1, MM / 128, HH), 256, 0, stream>>>(
          P_bf, MM, (long)MM * MM, vt_bf + (long)b * DD * MM, MM, (long)KH * MM,
          pa_f + boff, nullptr, DD, 32, 0,
          nullptr, nullptr, 0, MM, 1.0f, 1);
    }

    // residual projection: res = x @ W_res + b_res
    gemm_wmma_bf16<4><<<dim3(4, (BB * MM) / 128, 1), 256, 0, stream>>>(
        x_bf, DD, 0, wr_bf + i * DDsq, DD, 0,
        res_f, nullptr, DD, 0, 0, b_res + i * DD, nullptr, 0, DD, 1.0f, 0);

    // MLP: t1 = gelu(pa @ W1 + b1); x = gelu(t1 @ W2 + b2 + res)
    f32_to_bf16_k<<<(BB * MM * DD) / 256, 256, 0, stream>>>(pa_f, pa_bf, BB * MM * DD);
    gemm_wmma_bf16<4><<<dim3(4, (BB * MM) / 128, 1), 256, 0, stream>>>(
        pa_bf, DD, 0, w1_bf + i * DDsq, DD, 0,
        t1_f, nullptr, DD, 0, 0, b1_mlp + i * DD, nullptr, 0, DD, 1.0f, 1);
    f32_to_bf16_k<<<(BB * MM * DD) / 256, 256, 0, stream>>>(t1_f, t1_bf, BB * MM * DD);
    gemm_wmma_bf16<4><<<dim3(4, (BB * MM) / 128, 1), 256, 0, stream>>>(
        t1_bf, DD, 0, w2_bf + i * DDsq, DD, 0,
        x_f, nullptr, DD, 0, 0, b2_mlp + i * DD, res_f, DD, DD, 1.0f, 1);
  }

  // --- pos-att up: M=1024 in, N=4096 out; rank 306, frac 0.9 ---
  f32_to_bf16_k<<<(BB * MM * DD) / 256, 256, 0, stream>>>(x_f, x_bf, BB * MM * DD);
  gemm_wmma_bf16<4><<<dim3(4, (BB * MM) / 128, 1), 256, 0, stream>>>(
      x_bf, DD, 0, wup_bf, DD, 0,
      val_up, nullptr, DD, 0, 0, nullptr, nullptr, 0, DD, 1.0f, 0);
  pos_att_k<<<dim3(NN, HH), 256, 0, stream>>>(
      m_cross, hs, 8, val_up, de_f,
      MM, (long)MM, 1L, 306, 0.89999999999996589f,
      (long)MM * DD, (long)NN * DD);

  // --- decode: t = gelu(de @ W_de1 + b_de1); out = (t @ W_de2 + b)*std + mean ---
  f32_to_bf16_k<<<(BB * NN * DD) / 256, 256, 0, stream>>>(de_f, de_bf, BB * NN * DD);
  gemm_wmma_bf16<4><<<dim3(4, (BB * NN) / 128, 1), 256, 0, stream>>>(
      de_bf, DD, 0, wd1_bf, DD, 0,
      tde_f, nullptr, DD, 0, 0, b_de1, nullptr, 0, DD, 1.0f, 1);
  decode_final_k<<<(BB * NN) / 256, 256, 0, stream>>>(tde_f, W_de2, b_de2, y_mean, y_std, outp);
}
